// AdvancedNeuroCoder_7078106104163
// MI455X (gfx1250) — matrix operand order repaired and checked by reading, not compiled, not run on hardware
//
#include <hip/hip_runtime.h>
#include <stdint.h>

// ---------- types for WMMA ----------
typedef __attribute__((ext_vector_type(16))) __bf16 v16bf;
typedef __attribute__((ext_vector_type(16))) short  v16s;
typedef __attribute__((ext_vector_type(8)))  short  v8s;
typedef __attribute__((ext_vector_type(8)))  float  v8f;

union Frag16 { v16s s; v16bf b; v8s h[2]; };

// =====================================================================
// Batched WMMA GEMM:  C[M,N] = A[M,K](bf16) x B[K,N](bf16, stored transposed
// as Bt[N,K]) + bias, optional relu, f32 accumulate.
// Block = 256 threads = 8 waves computing a 128x64 macro-tile. The 64x32 bf16
// B panel (4 KB) is cooperatively staged into LDS and DOUBLE-BUFFERED. Issue
// order within a K-step: A-fragment loads first, THEN the next panel's
// staging load — VMEM retires in order, so the WMMAs can proceed on a partial
// loadcnt wait while the staging load is still in flight; it only needs to
// drain at the ds_store after the WMMAs. One barrier per K-step.
// Each wave: 16x64 strip, 4 v_wmma per K-step, A fragment held in registers.
// Requirements: M % 128 == 0, N % 64 == 0, K % 32 == 0, lda/ldb % 8 == 0.
// grid.y/grid.z provide batch offsets via element strides.
// =====================================================================
#define BPAD 40   // LDS row stride in shorts (80 B = 5*16 B, conflict-free)

__global__ __launch_bounds__(256) void wmma_gemm_bf16(
    const unsigned short* __restrict__ A, int lda, long long sAy, long long sAz,
    const unsigned short* __restrict__ Bt, int ldb, long long sBy, long long sBz,
    const float* __restrict__ bias,
    float* __restrict__ C, int ldc, long long sCy, long long sCz,
    int M, int N, int K, int relu)
{
    __shared__ unsigned short sB[2][64 * BPAD];

    const int lane = threadIdx.x & 31;
    const int wid  = threadIdx.x >> 5;
    const int half = lane >> 4;
    const int lr   = lane & 15;

    const int tn4   = N >> 6;                 // 64-wide column strips
    const int tile_m = blockIdx.x / tn4;      // 128-row macro tiles
    const int tile_n = blockIdx.x % tn4;

    A  += (long long)blockIdx.y * sAy + (long long)blockIdx.z * sAz;
    Bt += (long long)blockIdx.y * sBy + (long long)blockIdx.z * sBz;
    C  += (long long)blockIdx.y * sCy + (long long)blockIdx.z * sCz;

    const int row  = tile_m * 128 + wid * 16 + lr;   // this lane's A row
    const int col0 = tile_n * 64;

    // staging assignment: each thread copies one 16-byte chunk per K-step
    const int scol = threadIdx.x >> 2;        // 0..63
    const int skq  = threadIdx.x & 3;         // 0..3 (8 shorts each)
    const unsigned short* srow = Bt + (size_t)(col0 + scol) * ldb + skq * 8;
    unsigned short* sdst0 = &sB[0][scol * BPAD + skq * 8];
    unsigned short* sdst1 = &sB[1][scol * BPAD + skq * 8];

    v8f acc0 = {}, acc1 = {}, acc2 = {}, acc3 = {};
    const unsigned short* arow = A + (size_t)row * lda;

    // prologue: stage panel 0
    v8s stg = *reinterpret_cast<const v8s*>(srow);
    *reinterpret_cast<v8s*>(sdst0) = stg;
    __syncthreads();

    for (int k0 = 0; k0 < K; k0 += 32) {
        const int cb = (k0 >> 5) & 1;                       // consume buffer
        const unsigned short* sb = sB[cb];
        const bool more = (k0 + 32) < K;                    // block-uniform

        // A 16x32 bf16 fragment first: lanes 0-15 rows M=0..15 K[0..15],
        // lanes 16-31 same rows, K[16..31]
        Frag16 a;
        a.h[0] = *reinterpret_cast<const v8s*>(arow + k0 + half * 16);
        a.h[1] = *reinterpret_cast<const v8s*>(arow + k0 + half * 16 + 8);

        // next panel's staging load issued LAST: loads retire in order, so the
        // WMMAs below need only a partial loadcnt wait for the A data while
        // this one stays in flight until the ds_store after the WMMAs.
        if (more) stg = *reinterpret_cast<const v8s*>(srow + k0 + 32);

        // B 32x16 fragments from LDS: lane holds its column's 16 contiguous K
        Frag16 b0, b1, b2, b3;
        const v8s* p;
        p = reinterpret_cast<const v8s*>(&sb[(lr)      * BPAD + half * 16]); b0.h[0]=p[0]; b0.h[1]=p[1];
        p = reinterpret_cast<const v8s*>(&sb[(lr + 16) * BPAD + half * 16]); b1.h[0]=p[0]; b1.h[1]=p[1];
        p = reinterpret_cast<const v8s*>(&sb[(lr + 32) * BPAD + half * 16]); b2.h[0]=p[0]; b2.h[1]=p[1];
        p = reinterpret_cast<const v8s*>(&sb[(lr + 48) * BPAD + half * 16]); b3.h[0]=p[0]; b3.h[1]=p[1];

        acc0 = __builtin_amdgcn_wmma_f32_16x16x32_bf16(false, a.b, false, b0.b, (short)0, acc0, false, false);
        acc1 = __builtin_amdgcn_wmma_f32_16x16x32_bf16(false, a.b, false, b1.b, (short)0, acc1, false, false);
        acc2 = __builtin_amdgcn_wmma_f32_16x16x32_bf16(false, a.b, false, b2.b, (short)0, acc2, false, false);
        acc3 = __builtin_amdgcn_wmma_f32_16x16x32_bf16(false, a.b, false, b3.b, (short)0, acc3, false, false);

        // store next panel into the other buffer; readers of THIS buffer have
        // already pulled their fragments, writers touch only buffer cb^1, so a
        // single barrier per iteration is sufficient.
        if (more) *reinterpret_cast<v8s*>(cb ? sdst0 : sdst1) = stg;
        __syncthreads();
    }

    // C 16x16 f32: VGPR v -> row v + 8*half, col = lane%16
    const int rbase = tile_m * 128 + wid * 16 + half * 8;
    v8f accs[4] = {acc0, acc1, acc2, acc3};
    for (int nt = 0; nt < 4; ++nt) {
        int c = col0 + nt * 16 + lr;
        float bv = bias ? bias[c] : 0.0f;
        for (int v = 0; v < 8; ++v) {
            float val = accs[nt][v] + bv;
            if (relu) val = fmaxf(val, 0.0f);
            C[(size_t)(rbase + v) * ldc + c] = val;
        }
    }
}

// ---------- conversion kernels ----------
__device__ __forceinline__ unsigned short f2bf(float f) {
    unsigned int u = __float_as_uint(f);
    return (unsigned short)((u + 0x7fffu + ((u >> 16) & 1u)) >> 16);
}

__global__ void cvt_bf16(const float* __restrict__ in, unsigned short* __restrict__ out, long long n) {
    long long i = (long long)blockIdx.x * blockDim.x + threadIdx.x;
    if (i < n) out[i] = f2bf(in[i]);
}

// out[c*R + r] = bf16(in[r*C + c])   (produce [N,K] B-operand layout)
__global__ void cvt_bf16_t(const float* __restrict__ in, unsigned short* __restrict__ out, int R, int C) {
    long long i = (long long)blockIdx.x * blockDim.x + threadIdx.x;
    if (i >= (long long)R * C) return;
    int r = (int)(i / C), c = (int)(i % C);
    out[(size_t)c * R + r] = f2bf(in[i]);
}

// ---------- elementwise / small kernels ----------
__global__ void embed_k(const int* __restrict__ x, const float* __restrict__ emb,
                        float* __restrict__ out, long long n, int D) {
    long long i = (long long)blockIdx.x * blockDim.x + threadIdx.x;
    if (i >= n) return;
    long long bs = i / D; int d = (int)(i % D);
    out[i] = emb[(long long)x[bs] * D + d];
}

__global__ void fill0_k(float* p, long long n) {
    long long i = (long long)blockIdx.x * blockDim.x + threadIdx.x;
    if (i < n) p[i] = 0.0f;
}

__global__ void conv1x1_k(const float* __restrict__ in, const float* __restrict__ w,
                          const float* __restrict__ bias, float* __restrict__ out,
                          int Cin, int Cout, int ostride, int ooff, int Mtot) {
    int i = blockIdx.x * blockDim.x + threadIdx.x;
    if (i >= Mtot * Cout) return;
    int bs = i / Cout, o = i % Cout;
    const float* ip = in + (size_t)bs * Cin;
    const float* wp = w + (size_t)o * Cin;
    float acc = bias[o];
    for (int c = 0; c < Cin; ++c) acc += ip[c] * wp[c];
    out[(size_t)bs * ostride + ooff + o] = acc;
}

__global__ void convk_k(const float* __restrict__ in, const float* __restrict__ w,
                        const float* __restrict__ bias, float* __restrict__ out,
                        int Cin, int Cout, int Kw, int S, int ostride, int ooff, int Mtot) {
    int i = blockIdx.x * blockDim.x + threadIdx.x;
    if (i >= Mtot * Cout) return;
    int bs = i / Cout, o = i % Cout;
    int b = bs / S, s = bs % S;
    int pad = Kw / 2;
    float acc = bias[o];
    for (int t = 0; t < Kw; ++t) {
        int ss = s + t - pad;
        if (ss < 0 || ss >= S) continue;
        const float* ip = in + ((size_t)b * S + ss) * Cin;
        const float* wp = w + (size_t)o * Cin * Kw + t;
        for (int c = 0; c < Cin; ++c) acc += ip[c] * wp[(size_t)c * Kw];
    }
    out[(size_t)bs * ostride + ooff + o] = acc;
}

__global__ void maxpool3_k(const float* __restrict__ in, float* __restrict__ out, int S, int D, long long n) {
    long long i = (long long)blockIdx.x * blockDim.x + threadIdx.x;
    if (i >= n) return;
    long long bs = i / D;
    int s = (int)(bs % S);
    float m = in[i];
    if (s > 0)     m = fmaxf(m, in[i - D]);
    if (s < S - 1) m = fmaxf(m, in[i + D]);
    out[i] = m;
}

// quantum layer: h[row,0:768], two layers of sin-rotation + entangle, in place
__global__ void quantum_k(float* __restrict__ h, const float* __restrict__ q_rot,
                          const float* __restrict__ q_ent) {
    __shared__ float s0[768];
    __shared__ float s1[768];
    float* hp = h + (long long)blockIdx.x * 768;
    for (int i = threadIdx.x; i < 768; i += 256) s0[i] = hp[i];
    __syncthreads();
    for (int layer = 0; layer < 2; ++layer) {
        const float* rot = q_rot + (size_t)layer * 768 * 3;
        const float* ent = q_ent + (size_t)layer * 767;
        for (int q = threadIdx.x; q < 768; q += 256) {
            float v = s0[q];
            s1[q] = sinf(v + rot[q * 3]) + sinf(v + rot[q * 3 + 1]) + sinf(v + rot[q * 3 + 2]);
        }
        __syncthreads();
        for (int q = threadIdx.x; q < 768; q += 256) {
            if (q < 767) s0[q] = s1[q] * sinf(ent[q]) + s1[(q + 767) % 768] * cosf(ent[q]);
            else         s0[q] = 0.0f;
        }
        __syncthreads();
    }
    for (int i = threadIdx.x; i < 768; i += 256) hp[i] = s0[i];
}

// scale by 1/8, faithful mask bug (mask==1 -> -1e9), softmax over row of 1024
__global__ void softmax_k(float* __restrict__ scores, const int* __restrict__ mask, int S) {
    __shared__ float red[256];
    long long roff = (((long long)blockIdx.z * gridDim.y + blockIdx.y) * S + blockIdx.x) * (long long)S;
    float* row = scores + roff;
    const int* m = mask + (long long)blockIdx.z * S;
    float loc[4]; float mx = -1e30f;
    for (int i = 0; i < 4; ++i) {
        int c = threadIdx.x + i * 256;
        float v = (m[c] == 1) ? -1e9f : row[c] * 0.125f;
        loc[i] = v; mx = fmaxf(mx, v);
    }
    red[threadIdx.x] = mx; __syncthreads();
    for (int off = 128; off > 0; off >>= 1) {
        if (threadIdx.x < off) red[threadIdx.x] = fmaxf(red[threadIdx.x], red[threadIdx.x + off]);
        __syncthreads();
    }
    mx = red[0]; __syncthreads();
    float sum = 0.0f;
    for (int i = 0; i < 4; ++i) { loc[i] = __expf(loc[i] - mx); sum += loc[i]; }
    red[threadIdx.x] = sum; __syncthreads();
    for (int off = 128; off > 0; off >>= 1) {
        if (threadIdx.x < off) red[threadIdx.x] += red[threadIdx.x + off];
        __syncthreads();
    }
    float inv = 1.0f / red[0];
    for (int i = 0; i < 4; ++i) row[threadIdx.x + i * 256] = loc[i] * inv;
}

// out = LN(x + res) * g + b over D=768, one block per row
__global__ void ln_k(const float* __restrict__ x, const float* __restrict__ res,
                     const float* __restrict__ g, const float* __restrict__ bta,
                     float* __restrict__ out) {
    __shared__ float red[256];
    long long row = blockIdx.x;
    const float* xp = x + row * 768;
    const float* rp = res + row * 768;
    float v[3]; float s = 0.0f;
    for (int i = 0; i < 3; ++i) { int d = threadIdx.x + i * 256; v[i] = xp[d] + rp[d]; s += v[i]; }
    red[threadIdx.x] = s; __syncthreads();
    for (int off = 128; off > 0; off >>= 1) {
        if (threadIdx.x < off) red[threadIdx.x] += red[threadIdx.x + off];
        __syncthreads();
    }
    float mean = red[0] * (1.0f / 768.0f); __syncthreads();
    float s2 = 0.0f;
    for (int i = 0; i < 3; ++i) { float t = v[i] - mean; s2 += t * t; }
    red[threadIdx.x] = s2; __syncthreads();
    for (int off = 128; off > 0; off >>= 1) {
        if (threadIdx.x < off) red[threadIdx.x] += red[threadIdx.x + off];
        __syncthreads();
    }
    float inv = rsqrtf(red[0] * (1.0f / 768.0f) + 1e-5f);
    for (int i = 0; i < 3; ++i) {
        int d = threadIdx.x + i * 256;
        out[row * 768 + d] = (v[i] - mean) * inv * g[d] + bta[d];
    }
}

__global__ void mean_k(const float* __restrict__ h, float* __restrict__ out, int S, int D, int n) {
    int i = blockIdx.x * blockDim.x + threadIdx.x;
    if (i >= n) return;
    int b = i / D, d = i % D;
    float s = 0.0f;
    for (int ss = 0; ss < S; ++ss) s += h[((size_t)b * S + ss) * D + d];
    out[i] = s / (float)S;
}

__global__ void task_k(const float* __restrict__ hmean, const float* __restrict__ tw,
                       const float* __restrict__ tb, float* __restrict__ out) {
    int i = threadIdx.x;
    if (i >= 6) return;
    int b = i / 3, j = i % 3;
    float s = tb[j];
    for (int d = 0; d < 768; ++d) s += hmean[b * 768 + d] * tw[d * 3 + j];
    out[i] = s;
}

// =====================================================================
extern "C" void kernel_launch(void* const* d_in, const int* in_sizes, int n_in,
                              void* d_out, int out_size, void* d_ws, size_t ws_size,
                              hipStream_t stream) {
    const int B2 = 2, S = 1024, D = 768, V = 32000, H = 12, FF = 3072;
    const int M = B2 * S;                        // 2048 tokens
    (void)in_sizes; (void)n_in; (void)out_size; (void)ws_size;

    // ---- inputs (setup_inputs order) ----
    const int*   x      = (const int*)  d_in[0];
    const int*   amask  = (const int*)  d_in[1];
    const float* adj    = (const float*)d_in[2];
    const float* emb    = (const float*)d_in[3];
    const float* inc1_w = (const float*)d_in[4];  const float* inc1_b = (const float*)d_in[5];
    const float* inc3a_w= (const float*)d_in[6];  const float* inc3a_b= (const float*)d_in[7];
    const float* inc3b_w= (const float*)d_in[8];  const float* inc3b_b= (const float*)d_in[9];
    const float* inc5a_w= (const float*)d_in[10]; const float* inc5a_b= (const float*)d_in[11];
    const float* inc5b_w= (const float*)d_in[12]; const float* inc5b_b= (const float*)d_in[13];
    const float* incp_w = (const float*)d_in[14]; const float* incp_b = (const float*)d_in[15];
    const float* q_rot  = (const float*)d_in[16]; const float* q_ent  = (const float*)d_in[17];
    const float* gnn_w1 = (const float*)d_in[18]; const float* gnn_b1 = (const float*)d_in[19];
    const float* gnn_w2 = (const float*)d_in[20]; const float* gnn_b2 = (const float*)d_in[21];
    const float* wq = (const float*)d_in[22]; const float* bq = (const float*)d_in[23];
    const float* wk = (const float*)d_in[24]; const float* bk = (const float*)d_in[25];
    const float* wv = (const float*)d_in[26]; const float* bv = (const float*)d_in[27];
    const float* wo = (const float*)d_in[28]; const float* bo = (const float*)d_in[29];
    const float* ff_w1 = (const float*)d_in[30]; const float* ff_b1 = (const float*)d_in[31];
    const float* ff_w2 = (const float*)d_in[32]; const float* ff_b2 = (const float*)d_in[33];
    const float* ln1_g = (const float*)d_in[34]; const float* ln1_b = (const float*)d_in[35];
    const float* ln2_g = (const float*)d_in[36]; const float* ln2_b = (const float*)d_in[37];
    const float* out_w = (const float*)d_in[38]; const float* out_b = (const float*)d_in[39];
    const float* task_w= (const float*)d_in[40]; const float* task_b= (const float*)d_in[41];
    float* outp = (float*)d_out;

    // ---- workspace carving ----
    uintptr_t cur = (uintptr_t)d_ws;
    auto alloc = [&](size_t bytes) -> void* {
        uintptr_t r = (cur + 255) & ~(uintptr_t)255;
        cur = r + bytes;
        return (void*)r;
    };
    auto fbuf = [&](size_t n) { return (float*)alloc(n * sizeof(float)); };
    auto bbuf = [&](size_t n) { return (unsigned short*)alloc(n * sizeof(unsigned short)); };

    float* f_h0   = fbuf((size_t)M * D);
    float* f_hq   = fbuf((size_t)M * D);
    float* f_t96  = fbuf((size_t)M * 96);
    float* f_t16  = fbuf((size_t)M * 16);
    float* f_pool = fbuf((size_t)M * D);
    float* f_t1   = fbuf((size_t)M * D);
    float* f_g    = fbuf((size_t)M * 384);
    float* f_t2   = fbuf((size_t)M * 384);
    float* f_hg   = fbuf((size_t)M * D);
    float* f_Q    = fbuf((size_t)M * D);
    float* f_K    = fbuf((size_t)M * D);
    float* f_V    = fbuf((size_t)M * D);
    float* f_sc   = fbuf((size_t)B2 * H * S * S);
    float* f_ctx  = fbuf((size_t)M * D);
    float* f_ao   = fbuf((size_t)M * D);
    float* f_ln1  = fbuf((size_t)M * D);
    float* f_ff1  = fbuf((size_t)M * FF);
    float* f_ff2  = fbuf((size_t)M * D);
    float* f_ln2  = fbuf((size_t)M * D);
    float* f_hm   = fbuf((size_t)B2 * D);

    unsigned short* b_adj = bbuf((size_t)B2 * S * S);
    unsigned short* bt_h  = bbuf((size_t)D * M);
    unsigned short* b_t1  = bbuf((size_t)M * D);
    unsigned short* bt_g  = bbuf((size_t)384 * M);
    unsigned short* b_t2  = bbuf((size_t)M * 384);
    unsigned short* b_hg  = bbuf((size_t)M * D);
    unsigned short* b_w   = bbuf((size_t)V * D);          // shared weight slot (max out_w^T)
    unsigned short* b_Q   = bbuf((size_t)M * D);
    unsigned short* b_K   = bbuf((size_t)M * D);
    unsigned short* b_at  = bbuf((size_t)B2 * H * S * S);
    unsigned short* bt_V  = bbuf((size_t)D * M);
    unsigned short* b_ctx = bbuf((size_t)M * D);
    unsigned short* b_ln1 = bbuf((size_t)M * D);
    unsigned short* b_ff1 = bbuf((size_t)M * FF);
    unsigned short* b_ln2 = bbuf((size_t)M * D);

    auto gblk = [](long long n) { return (unsigned)((n + 255) / 256); };
    auto cvt  = [&](const float* in, unsigned short* out, long long n) {
        hipLaunchKernelGGL(cvt_bf16, dim3(gblk(n)), dim3(256), 0, stream, in, out, n);
    };
    auto cvtT = [&](const float* in, unsigned short* out, int R, int C) {
        hipLaunchKernelGGL(cvt_bf16_t, dim3(gblk((long long)R * C)), dim3(256), 0, stream, in, out, R, C);
    };
    auto gemm = [&](const unsigned short* A, int lda, long long sAy, long long sAz,
                    const unsigned short* Bt, int ldb, long long sBy, long long sBz,
                    const float* bias, float* C, int ldc, long long sCy, long long sCz,
                    int Mi, int Ni, int Ki, int relu, int gy, int gz) {
        dim3 grid((unsigned)((Mi / 128) * (Ni / 64)), (unsigned)gy, (unsigned)gz);
        hipLaunchKernelGGL(wmma_gemm_bf16, grid, dim3(256), 0, stream,
                           A, lda, sAy, sAz, Bt, ldb, sBy, sBz, bias,
                           C, ldc, sCy, sCz, Mi, Ni, Ki, relu);
    };

    const long long nMD = (long long)M * D;

    // 1) embedding
    hipLaunchKernelGGL(embed_k, dim3(gblk(nMD)), dim3(256), 0, stream, x, emb, f_h0, nMD, D);

    // 2) inception branches -> f_hq channels [b1|b3|b5|bp|zeros] (NQ = 768)
    hipLaunchKernelGGL(fill0_k, dim3(gblk(nMD)), dim3(256), 0, stream, f_hq, nMD);
    hipLaunchKernelGGL(conv1x1_k, dim3(gblk((long long)M * 64)), dim3(256), 0, stream,
                       f_h0, inc1_w, inc1_b, f_hq, D, 64, D, 0, M);
    hipLaunchKernelGGL(conv1x1_k, dim3(gblk((long long)M * 96)), dim3(256), 0, stream,
                       f_h0, inc3a_w, inc3a_b, f_t96, D, 96, 96, 0, M);
    hipLaunchKernelGGL(convk_k, dim3(gblk((long long)M * 128)), dim3(256), 0, stream,
                       f_t96, inc3b_w, inc3b_b, f_hq, 96, 128, 3, S, D, 64, M);
    hipLaunchKernelGGL(conv1x1_k, dim3(gblk((long long)M * 16)), dim3(256), 0, stream,
                       f_h0, inc5a_w, inc5a_b, f_t16, D, 16, 16, 0, M);
    hipLaunchKernelGGL(convk_k, dim3(gblk((long long)M * 32)), dim3(256), 0, stream,
                       f_t16, inc5b_w, inc5b_b, f_hq, 16, 32, 5, S, D, 192, M);
    hipLaunchKernelGGL(maxpool3_k, dim3(gblk(nMD)), dim3(256), 0, stream, f_h0, f_pool, S, D, nMD);
    hipLaunchKernelGGL(conv1x1_k, dim3(gblk((long long)M * 32)), dim3(256), 0, stream,
                       f_pool, incp_w, incp_b, f_hq, D, 32, D, 224, M);

    // 3) quantum layer (in place on f_hq)
    hipLaunchKernelGGL(quantum_k, dim3(M), dim3(256), 0, stream, f_hq, q_rot, q_ent);

    // 4) GNN: t1 = adj @ h ; g = relu(t1 @ w1 + b1); t2 = adj @ g; hg = t2 @ w2 + b2
    cvt(adj, b_adj, (long long)B2 * S * S);
    cvtT(f_hq, bt_h, M, D);                                   // [D, M]
    gemm(b_adj, S, 0, (long long)S * S,
         bt_h, M, 0, S,
         nullptr, f_t1, D, 0, (long long)S * D,
         S, D, S, 0, 1, B2);
    cvt(f_t1, b_t1, nMD);
    cvtT(gnn_w1, b_w, D, 384);                                 // [384, D]
    gemm(b_t1, D, 0, 0, b_w, D, 0, 0, gnn_b1, f_g, 384, 0, 0, M, 384, D, 1, 1, 1);
    cvtT(f_g, bt_g, M, 384);                                   // [384, M]
    gemm(b_adj, S, 0, (long long)S * S,
         bt_g, M, 0, S,
         nullptr, f_t2, 384, 0, (long long)S * 384,
         S, 384, S, 0, 1, B2);
    cvt(f_t2, b_t2, (long long)M * 384);
    cvtT(gnn_w2, b_w, 384, D);                                 // [D, 384]
    gemm(b_t2, 384, 0, 0, b_w, 384, 0, 0, gnn_b2, f_hg, D, 0, 0, M, D, 384, 0, 1, 1);

    // 5) attention projections
    cvt(f_hg, b_hg, nMD);
    cvtT(wq, b_w, D, D);
    gemm(b_hg, D, 0, 0, b_w, D, 0, 0, bq, f_Q, D, 0, 0, M, D, D, 0, 1, 1);
    cvtT(wk, b_w, D, D);
    gemm(b_hg, D, 0, 0, b_w, D, 0, 0, bk, f_K, D, 0, 0, M, D, D, 0, 1, 1);
    cvtT(wv, b_w, D, D);
    gemm(b_hg, D, 0, 0, b_w, D, 0, 0, bv, f_V, D, 0, 0, M, D, D, 0, 1, 1);

    // 6) scores = Q K^T per (b,h); K natural [M,D] layout IS the transposed B operand
    cvt(f_Q, b_Q, nMD);
    cvt(f_K, b_K, nMD);
    gemm(b_Q, D, 64, (long long)S * D,
         b_K, D, 64, (long long)S * D,
         nullptr, f_sc, S, (long long)S * S, (long long)H * S * S,
         S, S, 64, 0, H, B2);

    // 7) scale + mask(+bug) + softmax, in place
    {
        dim3 grid((unsigned)S, (unsigned)H, (unsigned)B2);
        hipLaunchKernelGGL(softmax_k, grid, dim3(256), 0, stream, f_sc, amask, S);
    }

    // 8) ctx = attn @ V per (b,h)
    cvt(f_sc, b_at, (long long)B2 * H * S * S);
    cvtT(f_V, bt_V, M, D);                                     // [D, M]
    gemm(b_at, S, (long long)S * S, (long long)H * S * S,
         bt_V, M, (long long)64 * M, S,
         nullptr, f_ctx, D, 64, (long long)S * D,
         S, 64, S, 0, H, B2);

    // 9) output projection + residual LN1
    cvt(f_ctx, b_ctx, nMD);
    cvtT(wo, b_w, D, D);
    gemm(b_ctx, D, 0, 0, b_w, D, 0, 0, bo, f_ao, D, 0, 0, M, D, D, 0, 1, 1);
    hipLaunchKernelGGL(ln_k, dim3(M), dim3(256), 0, stream, f_hg, f_ao, ln1_g, ln1_b, f_ln1);

    // 10) FFN + residual LN2
    cvt(f_ln1, b_ln1, nMD);
    cvtT(ff_w1, b_w, D, FF);                                   // [FF, D]
    gemm(b_ln1, D, 0, 0, b_w, D, 0, 0, ff_b1, f_ff1, FF, 0, 0, M, FF, D, 1, 1, 1);
    cvt(f_ff1, b_ff1, (long long)M * FF);
    cvtT(ff_w2, b_w, FF, D);                                   // [D, FF]
    gemm(b_ff1, FF, 0, 0, b_w, FF, 0, 0, ff_b2, f_ff2, D, 0, 0, M, D, FF, 0, 1, 1);
    hipLaunchKernelGGL(ln_k, dim3(M), dim3(256), 0, stream, f_ln1, f_ff2, ln2_g, ln2_b, f_ln2);

    // 11) token logits: [2048, 32000] straight into d_out (the 100-GFLOP GEMM)
    cvt(f_ln2, b_ln2, nMD);
    cvtT(out_w, b_w, D, V);                                    // [V, D]
    gemm(b_ln2, D, 0, 0, b_w, D, 0, 0, out_b, outp, V, 0, 0, M, V, D, 0, 1, 1);

    // 12) task head: mean over S then tiny GEMM, appended after token output
    hipLaunchKernelGGL(mean_k, dim3(gblk(B2 * D)), dim3(256), 0, stream, f_ln2, f_hm, S, D, B2 * D);
    hipLaunchKernelGGL(task_k, dim3(1), dim3(32), 0, stream, f_hm, task_w, task_b,
                       outp + (long long)B2 * S * V);
}
